// FusedBNReLUConv_33921651704500
// MI455X (gfx1250) — matrix operand order repaired and checked by reading, not compiled, not run on hardware
//
#include <hip/hip_runtime.h>
#include <hip/hip_bf16.h>

// CDNA5 / gfx1250, wave32. Implicit-GEMM fused BN+ReLU+3x3 conv using
// v_wmma_f32_16x16x32_bf16 (bf16 A/B, f32 accumulate).
// 4 output rows per block => x halo traffic 1.5x instead of 3x (memory-bound
// kernel; HBM traffic is the dominant cost at 23.3 TB/s).

typedef __attribute__((ext_vector_type(16))) __bf16 v16bf;
typedef __attribute__((ext_vector_type(8)))  float  v8f;

union BfVec {
    v16bf v;
    uint4 q[2];
};

__device__ __forceinline__ unsigned bf16_rne(float f) {
    unsigned u = __float_as_uint(f);
    u = (u + 0x7FFFu + ((u >> 16) & 1u)) >> 16;
    return u & 0xFFFFu;
}
__device__ __forceinline__ unsigned pack2_bf16(float lo, float hi) {
    return bf16_rne(lo) | (bf16_rne(hi) << 16);
}

// ---------------------------------------------------------------------------
// Weight prep: fp32 HWIO [3][3][768][32]  ->  bf16, pre-swizzled into the
// exact WMMA B-operand per-lane layout.
// Region r = ((ky*3+kx)*24 + ct)*2 + co  (ct = 32-channel chunk, co = cout half)
// Within a region: lane (0..31) stores its 16 bf16 B elements contiguously:
//   column  n  = co*16 + (lane & 15)
//   K rows     = ct*32 + (lane>=16 ? 16 : 0) + j,  j = 0..15
// => main kernel loads each B tile with two global b128 per lane.
// ---------------------------------------------------------------------------
__global__ void prep_weights_kernel(const float* __restrict__ w,
                                    unsigned short* __restrict__ wq) {
    int id = blockIdx.x * blockDim.x + threadIdx.x;
    if (id >= 432 * 512) return;
    int r    = id >> 9;          // region
    int rem  = id & 511;
    int lane = rem >> 4;
    int j    = rem & 15;
    int co   = r & 1;
    int ct   = (r >> 1) % 24;
    int kxy  = r / 48;           // ky*3 + kx
    int c    = ct * 32 + ((lane >> 4) << 4) + j;
    int cout = co * 16 + (lane & 15);
    float f  = w[(kxy * 768 + c) * 32 + cout];
    wq[id] = (unsigned short)bf16_rne(f);
}

// ---------------------------------------------------------------------------
// Main kernel. One block per (n, h-group of 4 rows): 4 x 64 positions x 32
// couts. 8 waves: wave = (cout_half << 2) | wseg ; each wave owns one 16-wide
// column segment across the 4 output rows => 4 accumulator tiles, with the B
// (weight) tile reused across all 4 WMMAs of each tap.
// ---------------------------------------------------------------------------
__global__ __launch_bounds__(256) void fused_bn_relu_conv3x3_kernel(
    const float* __restrict__ x,
    const float* __restrict__ a_bn,
    const float* __restrict__ b_bn,
    const uint4* __restrict__ wq,   // pre-swizzled bf16 weights
    float* __restrict__ out) {

    // bf16 activation tile: [6 rows][66 cols][32 ch] = 25344 B
    __shared__ uint4 lds[1584];

    const int b    = blockIdx.x;       // n*16 + h/4
    const int n    = b >> 4;
    const int h0   = (b & 15) * 4;     // first output row of this block
    const int tid  = threadIdx.x;
    const int lane = tid & 31;
    const int wave = tid >> 5;
    const int pq   = wave & 3;         // column segment (16 positions)
    const int co   = wave >> 2;        // cout half (16 couts)

    v8f acc[4] = {{}, {}, {}, {}};

    // Per-lane loop-invariant A base (uint4 index): column within tile row.
    const int col0  = pq * 16 + (lane & 15);
    const int abase = col0 * 4 + (lane >> 4) * 2;

    for (int ct = 0; ct < 24; ++ct) {
        // ---- stage: BN affine + ReLU + cvt bf16 into LDS ----------------
        for (int q = tid; q < 396; q += 256) {     // 6 rows x 66 cols
            const int r6   = q / 66;
            const int colt = q - r6 * 66;
            const int row  = h0 - 1 + r6;          // image row
            const int col  = colt - 1;             // image col
            unsigned tmp[16];
            if ((unsigned)row < 64u && (unsigned)col < 64u) {
                const float4* x4 = (const float4*)(x + (((n * 64 + row) * 64 + col) * 768 + ct * 32));
                const float4* a4 = (const float4*)(a_bn + ct * 32);
                const float4* b4 = (const float4*)(b_bn + ct * 32);
#pragma unroll
                for (int j = 0; j < 8; ++j) {
                    float4 xv = x4[j], av = a4[j], bv = b4[j];
                    float f0 = fmaxf(fmaf(xv.x, av.x, bv.x), 0.0f);
                    float f1 = fmaxf(fmaf(xv.y, av.y, bv.y), 0.0f);
                    float f2 = fmaxf(fmaf(xv.z, av.z, bv.z), 0.0f);
                    float f3 = fmaxf(fmaf(xv.w, av.w, bv.w), 0.0f);
                    tmp[j * 2 + 0] = pack2_bf16(f0, f1);
                    tmp[j * 2 + 1] = pack2_bf16(f2, f3);
                }
            } else {
#pragma unroll
                for (int j = 0; j < 16; ++j) tmp[j] = 0u;  // SAME padding
            }
#pragma unroll
            for (int j = 0; j < 4; ++j)
                lds[q * 4 + j] = make_uint4(tmp[j * 4 + 0], tmp[j * 4 + 1],
                                            tmp[j * 4 + 2], tmp[j * 4 + 3]);
        }
        __syncthreads();

        // ---- compute: 9 taps x 4 row-tiles, B reused per tap ------------
#pragma unroll
        for (int ky = 0; ky < 3; ++ky) {
#pragma unroll
            for (int kx = 0; kx < 3; ++kx) {
                BfVec B;
                const int reg = ((ky * 3 + kx) * 24 + ct) * 2 + co;
                const uint4* bp = wq + reg * 64 + lane * 2;
                B.q[0] = bp[0];
                B.q[1] = bp[1];
#pragma unroll
                for (int j = 0; j < 4; ++j) {
                    // A: row m = col0, tile row = j+ky, tile col = col0+kx
                    BfVec A;
                    const int ai = abase + ((j + ky) * 66 + kx) * 4;
                    A.q[0] = lds[ai];       // K 0..7  (or 16..23 upper lanes)
                    A.q[1] = lds[ai + 1];   // K 8..15 (or 24..31)
                    acc[j] = __builtin_amdgcn_wmma_f32_16x16x32_bf16(
                        false, A.v, false, B.v, (short)0, acc[j], false, false);
                }
            }
        }
        __syncthreads();
    }

    // ---- store D: lane = cout column, VGPR r = row r (+8 upper lanes) ---
    const int cout = co * 16 + (lane & 15);
    const int mofs = (lane >> 4) * 8;
#pragma unroll
    for (int j = 0; j < 4; ++j) {
        float* op = out + (((n * 64 + h0 + j) * 64) + pq * 16 + mofs) * 32 + cout;
#pragma unroll
        for (int r = 0; r < 8; ++r) op[r * 32] = acc[j][r];
    }
}

// ---------------------------------------------------------------------------
extern "C" void kernel_launch(void* const* d_in, const int* in_sizes, int n_in,
                              void* d_out, int out_size, void* d_ws, size_t ws_size,
                              hipStream_t stream) {
    const float* x    = (const float*)d_in[0];   // [16,64,64,768]
    const float* w    = (const float*)d_in[1];   // [3,3,768,32] HWIO
    const float* a_bn = (const float*)d_in[2];   // [768]
    const float* b_bn = (const float*)d_in[3];   // [768]
    float* out = (float*)d_out;                  // [16,64,64,32]

    unsigned short* wq = (unsigned short*)d_ws;  // 432 KiB bf16 swizzled weights

    prep_weights_kernel<<<(432 * 512 + 255) / 256, 256, 0, stream>>>(w, wq);
    fused_bn_relu_conv3x3_kernel<<<16 * 16, 256, 0, stream>>>(
        x, a_bn, b_bn, (const uint4*)wq, out);
}
